// MapLatentProcessor_46806553592104
// MI455X (gfx1250) — compile-verified, matrix operand order checked by reading, not run on hardware
//
#include <hip/hip_runtime.h>
#include <hip/hip_bf16.h>

// ---------------------------------------------------------------------------
// MapLatentProcessor for MI455X (gfx1250): bf16 WMMA everywhere, fp32 residual.
// GEMM register-blocked 4x2 tiles/wave (64x32 per wave) for WMMA density.
// ---------------------------------------------------------------------------

#define DEV __device__ __forceinline__

typedef __attribute__((ext_vector_type(4)))  unsigned int v4u;
typedef __attribute__((ext_vector_type(8)))  float        v8f;
typedef __attribute__((ext_vector_type(16))) __bf16       v16bf;

union FragU { v4u q[2]; v16bf v; };

DEV unsigned short f2bf(float f) {
  unsigned int u = __float_as_uint(f);
  u += 0x7FFFu + ((u >> 16) & 1u);            // round-to-nearest-even
  return (unsigned short)(u >> 16);
}

// Load a 16x32 bf16 fragment (A layout) or a [N,K]-major B fragment.
// Element (m|n, k): base[(row + (lane&15))*ld + kb + k], per-lane K split:
// lanes 0-15 hold K = {kb..kb+7, kb+16..kb+23}; lanes 16-31 the +8 halves.
DEV FragU load_frag(const unsigned short* base, int row, int ld, int kb, int lane) {
  int m = lane & 15, g = lane >> 4;
  const unsigned short* p = base + (size_t)(row + m) * ld + kb + 8 * g;
  FragU f;
  f.q[0] = *reinterpret_cast<const v4u*>(p);
  f.q[1] = *reinterpret_cast<const v4u*>(p + 16);
  return f;
}

DEV v8f wmma_bf16(const FragU& a, const FragU& b, v8f c) {
  return __builtin_amdgcn_wmma_f32_16x16x32_bf16(false, a.v, false, b.v,
                                                 (short)0, c, false, false);
}

// ---------------------------------------------------------------------------
// Register-blocked WMMA GEMM: C[M,N] = A[M,K](bf16) * W[N,K](bf16)^T (+bias)
// Each wave: 4 M-tiles x 2 N-tiles (64x32 block), 8 WMMAs per K-step from
// 6 fragment loads. MODE 0: store f32  1: += f32  2: store bf16  3: gelu->bf16
// ---------------------------------------------------------------------------
template<int MODE>
__global__ void gemm_bf16_kernel(const unsigned short* __restrict__ A, int lda,
                                 const unsigned short* __restrict__ W, int ldw,
                                 const float* __restrict__ bias,
                                 void* __restrict__ Out, int ldo,
                                 int Mb, int Nb, int K) {
  int wid = blockIdx.x * (blockDim.x >> 5) + (threadIdx.x >> 5);
  if (wid >= Mb * Nb) return;
  int lane = threadIdx.x & 31;
  int row0 = (wid / Nb) * 64;
  int col0 = (wid % Nb) * 32;

  v8f acc[4][2];
#pragma unroll
  for (int t = 0; t < 4; ++t)
#pragma unroll
    for (int j = 0; j < 2; ++j)
      acc[t][j] = (v8f){0.f,0.f,0.f,0.f,0.f,0.f,0.f,0.f};

  int m = lane & 15, g = lane >> 4;
  for (int kb = 0; kb < K; kb += 32) {
    if (kb + 32 < K) {
      // prefetch next K-step of the streaming A rows (global_prefetch_b8)
      __builtin_prefetch((const void*)(A + (size_t)(row0 + m) * lda + kb + 32 + 8 * g), 0, 1);
      __builtin_prefetch((const void*)(A + (size_t)(row0 + 32 + m) * lda + kb + 32 + 8 * g), 0, 1);
    }
    FragU b0 = load_frag(W, col0,      ldw, kb, lane);
    FragU b1 = load_frag(W, col0 + 16, ldw, kb, lane);
    FragU a0 = load_frag(A, row0,      lda, kb, lane);
    FragU a1 = load_frag(A, row0 + 16, lda, kb, lane);
    FragU a2 = load_frag(A, row0 + 32, lda, kb, lane);
    FragU a3 = load_frag(A, row0 + 48, lda, kb, lane);
    acc[0][0] = wmma_bf16(a0, b0, acc[0][0]);
    acc[0][1] = wmma_bf16(a0, b1, acc[0][1]);
    acc[1][0] = wmma_bf16(a1, b0, acc[1][0]);
    acc[1][1] = wmma_bf16(a1, b1, acc[1][1]);
    acc[2][0] = wmma_bf16(a2, b0, acc[2][0]);
    acc[2][1] = wmma_bf16(a2, b1, acc[2][1]);
    acc[3][0] = wmma_bf16(a3, b0, acc[3][0]);
    acc[3][1] = wmma_bf16(a3, b1, acc[3][1]);
  }

  int n = lane & 15;
  float bv0 = bias ? bias[col0 + n]      : 0.0f;
  float bv1 = bias ? bias[col0 + 16 + n] : 0.0f;
#pragma unroll
  for (int t = 0; t < 4; ++t) {
#pragma unroll
    for (int j = 0; j < 2; ++j) {
      float bv = j ? bv1 : bv0;
#pragma unroll
      for (int r = 0; r < 8; ++r) {
        size_t oi = (size_t)(row0 + t * 16 + r + 8 * g) * ldo + (col0 + j * 16 + n);
        float v = acc[t][j][r] + bv;
        if (MODE == 0)      ((float*)Out)[oi] = v;
        else if (MODE == 1) ((float*)Out)[oi] += v;
        else if (MODE == 2) ((unsigned short*)Out)[oi] = f2bf(v);
        else {
          float ge = 0.5f * v * (1.0f + erff(v * 0.70710678118f)); // exact gelu
          ((unsigned short*)Out)[oi] = f2bf(ge);
        }
      }
    }
  }
}

// ---------------------------------------------------------------------------
// Fused attention: one block per (batch, head, 32-query tile). S=320, dh=32.
// QK^T via WMMA -> f32 scores in LDS -> softmax -> bf16 probs -> PV via WMMA.
// LDS: 32*328*4 + 32*320*2 = 62464 B.
// ---------------------------------------------------------------------------
__global__ void attn_kernel(const unsigned short* __restrict__ qkv,
                            unsigned short* __restrict__ o) {
  constexpr int S = 320, H = 8, QT = 32, NQT = S / QT;
  __shared__ __align__(16) float          sc[QT][S + 8];
  __shared__ __align__(16) unsigned short pb[QT][S];
  int b  = blockIdx.x / (H * NQT);
  int h  = (blockIdx.x / NQT) % H;
  int qt = blockIdx.x % NQT;
  int rowbase = b * S;
  int qbase = qt * QT;
  int wave = threadIdx.x >> 5, lane = threadIdx.x & 31;
  int g = lane >> 4, n15 = lane & 15;
  const float scale = 0.17677669529663687f;    // 1/sqrt(32)

  // Phase 1: scores (2 M-tiles x 20 N-tiles = 40 tiles, 5 per wave)
#pragma unroll
  for (int i = 0; i < 5; ++i) {
    int t = wave * 5 + i;
    int mt = t % 2, nt = t / 2;
    FragU a  = load_frag(qkv, rowbase + qbase + mt * 16, 768, h * 32, lane);
    FragU bb = load_frag(qkv, rowbase + nt * 16, 768, 256 + h * 32, lane);
    v8f c = {0.f,0.f,0.f,0.f,0.f,0.f,0.f,0.f};
    c = wmma_bf16(a, bb, c);
#pragma unroll
    for (int r = 0; r < 8; ++r)
      sc[mt * 16 + r + 8 * g][nt * 16 + n15] = c[r] * scale;
  }
  __syncthreads();

  // Phase 2: softmax, 4 rows per wave (wave32 shuffle reductions)
  for (int rr = wave * 4; rr < wave * 4 + 4; ++rr) {
    float mx = -1e30f;
    for (int c0 = lane; c0 < S; c0 += 32) mx = fmaxf(mx, sc[rr][c0]);
    for (int d = 16; d; d >>= 1) mx = fmaxf(mx, __shfl_xor(mx, d, 32));
    float sum = 0.f;
    for (int c0 = lane; c0 < S; c0 += 32) {
      float e = __expf(sc[rr][c0] - mx);
      sc[rr][c0] = e; sum += e;
    }
    for (int d = 16; d; d >>= 1) sum += __shfl_xor(sum, d, 32);
    float inv = 1.0f / sum;
    for (int c0 = lane; c0 < S; c0 += 32) pb[rr][c0] = f2bf(sc[rr][c0] * inv);
  }
  __syncthreads();

  // Phase 3a: V^T [32][320] overlaid on the (now dead) score buffer
  unsigned short* vT = reinterpret_cast<unsigned short*>(&sc[0][0]);
  for (int e = threadIdx.x; e < 32 * S; e += blockDim.x) {
    int d = e / S, s = e % S;
    vT[d * S + s] = qkv[(size_t)(rowbase + s) * 768 + 512 + h * 32 + d];
  }
  __syncthreads();

  // Phase 3b: out[32x32] = probs[32x320] @ V[320x32]; 4 tiles, waves 0..3
  if (wave < 4) {
    int mt = wave & 1, nt = wave >> 1;
    v8f c = {0.f,0.f,0.f,0.f,0.f,0.f,0.f,0.f};
    for (int kb = 0; kb < S; kb += 32) {
      FragU a  = load_frag(&pb[0][0], mt * 16, S, kb, lane);
      FragU bb = load_frag(vT, nt * 16, S, kb, lane);
      c = wmma_bf16(a, bb, c);
    }
#pragma unroll
    for (int r = 0; r < 8; ++r)
      o[(size_t)(rowbase + qbase + mt * 16 + r + 8 * g) * 256 +
        h * 32 + nt * 16 + n15] = f2bf(c[r]);
  }
}

// ---------------------------------------------------------------------------
// LayerNorm over D=256, one wave per row (8 rows per 256-thread block).
// OUTMODE 0: f32, 1: bf16
// ---------------------------------------------------------------------------
template<int OUTMODE>
__global__ void ln_kernel(const float* __restrict__ x, const float* __restrict__ gw,
                          const float* __restrict__ bw, void* __restrict__ out,
                          int nrows) {
  int row = blockIdx.x * (blockDim.x >> 5) + (threadIdx.x >> 5);
  if (row >= nrows) return;
  int lane = threadIdx.x & 31;
  const float* xr = x + (size_t)row * 256;
  float v[8]; float s = 0.f;
#pragma unroll
  for (int i = 0; i < 8; ++i) { v[i] = xr[lane + 32 * i]; s += v[i]; }
  for (int d = 16; d; d >>= 1) s += __shfl_xor(s, d, 32);
  float mu = s * (1.0f / 256.0f);
  float q = 0.f;
#pragma unroll
  for (int i = 0; i < 8; ++i) { float t = v[i] - mu; q += t * t; }
  for (int d = 16; d; d >>= 1) q += __shfl_xor(q, d, 32);
  float rstd = rsqrtf(q * (1.0f / 256.0f) + 1e-5f);
#pragma unroll
  for (int i = 0; i < 8; ++i) {
    int cidx = lane + 32 * i;
    float y = (v[i] - mu) * rstd * gw[cidx] + bw[cidx];
    if (OUTMODE == 0) ((float*)out)[(size_t)row * 256 + cidx] = y;
    else ((unsigned short*)out)[(size_t)row * 256 + cidx] = f2bf(y);
  }
}

// ---------------------------------------------------------------------------
// Elementwise / data-movement kernels
// ---------------------------------------------------------------------------
DEV float sincos_pe(int d, int py, int px) {
  int dd = d & 127;
  float p = (d < 128) ? (float)py : (float)px;
  int i = dd >> 1;
  float div = __expf((float)(2 * i) * (-9.210340371976184f / 128.0f));
  float ang = p * div;
  return (dd & 1) ? __cosf(ang) : __sinf(ang);
}

__global__ void cvt_kernel(const float* __restrict__ src,
                           unsigned short* __restrict__ dst, size_t n) {
  size_t i = (size_t)blockIdx.x * blockDim.x + threadIdx.x;
  if (i < n) dst[i] = f2bf(src[i]);
}

__global__ void build_tokens_kernel(const float* __restrict__ hmap,
                                    const int* __restrict__ idx,
                                    const float* __restrict__ patch_b,
                                    float* __restrict__ xf,
                                    float* __restrict__ hcv) {
  size_t gid = (size_t)blockIdx.x * blockDim.x + threadIdx.x;
  if (gid >= (size_t)64 * 320 * 256) return;
  int d = (int)(gid & 255);
  size_t bt = gid >> 8;
  int t = (int)(bt % 320);
  int b = (int)(bt / 320);
  float base, pos;
  if (t < 256) {
    int s = idx[t];
    pos = sincos_pe(d, s / 33, s % 33);
    float hv = hmap[((size_t)b * 1089 + s) * 256 + d];
    hcv[((size_t)b * 256 + t) * 256 + d] = hv;
    base = hv;
  } else {
    int hw = t - 256;
    pos = sincos_pe(d, hw >> 3, hw & 7);
    base = patch_b[d];
  }
  xf[gid] = base + pos;
}

__global__ void build_patchA_kernel(const float* __restrict__ frame,
                                    unsigned short* __restrict__ A) {
  size_t gid = (size_t)blockIdx.x * blockDim.x + threadIdx.x;
  if (gid >= (size_t)4096 * 768) return;
  int k = (int)(gid % 768);
  size_t t = gid / 768;
  int hw = (int)(t & 63), b = (int)(t >> 6);
  int cch = k >> 8, p = (k >> 4) & 15, q = k & 15;
  int row = (hw >> 3) * 16 + p, col = (hw & 7) * 16 + q;
  A[gid] = f2bf(frame[(((size_t)b * 3 + cch) * 128 + row) * 128 + col]);
}

__global__ void add_patch_kernel(const float* __restrict__ tmp,
                                 float* __restrict__ xf) {
  size_t gid = (size_t)blockIdx.x * blockDim.x + threadIdx.x;
  if (gid >= (size_t)4096 * 256) return;
  int d = (int)(gid & 255);
  size_t t = gid >> 8;
  int hw = (int)(t & 63), b = (int)(t >> 6);
  xf[((size_t)b * 320 + 256 + hw) * 256 + d] += tmp[gid];
}

__global__ void build_cat_kernel(const float* __restrict__ hcv,
                                 const float* __restrict__ xnf,
                                 unsigned short* __restrict__ cat) {
  size_t gid = (size_t)blockIdx.x * blockDim.x + threadIdx.x;
  if (gid >= (size_t)16384 * 512) return;
  int c = (int)(gid & 511);
  size_t t = gid >> 9;                 // t = b*256 + r
  int r = (int)(t & 255), b = (int)(t >> 8);
  float v = (c < 256) ? hcv[t * 256 + c]
                      : xnf[((size_t)b * 320 + r) * 256 + (c - 256)];
  cat[gid] = f2bf(v);
}

__global__ void copy_kernel(const float* __restrict__ src,
                            float* __restrict__ dst, size_t n) {
  size_t i = (size_t)blockIdx.x * blockDim.x + threadIdx.x;
  if (i < n) dst[i] = src[i];
}

__global__ void blend_kernel(const float* __restrict__ gbuf,
                             const float* __restrict__ hcv,
                             const float* __restrict__ xnf,
                             const int* __restrict__ idx,
                             float* __restrict__ out) {
  size_t gid = (size_t)blockIdx.x * blockDim.x + threadIdx.x;
  if (gid >= (size_t)16384 * 256) return;
  int d = (int)(gid & 255);
  size_t t = gid >> 8;                 // t = b*256 + r
  int r = (int)(t & 255), b = (int)(t >> 8);
  float gg = 1.0f / (1.0f + __expf(-gbuf[gid]));
  float a = hcv[gid];
  float nn = xnf[((size_t)b * 320 + r) * 256 + d];
  out[((size_t)b * 1089 + idx[r]) * 256 + d] = (1.0f - gg) * a + gg * nn;
}

// ---------------------------------------------------------------------------
extern "C" void kernel_launch(void* const* d_in, const int* in_sizes, int n_in,
                              void* d_out, int out_size, void* d_ws, size_t ws_size,
                              hipStream_t stream) {
  const float* hmap    = (const float*)d_in[0];
  const float* frame   = (const float*)d_in[1];
  const int*   idx     = (const int*)  d_in[2];
  const float* Wqkv    = (const float*)d_in[3];
  const float* bqkv    = (const float*)d_in[4];
  const float* Wo      = (const float*)d_in[5];
  const float* bo      = (const float*)d_in[6];
  const float* ln1g    = (const float*)d_in[7];
  const float* ln1b    = (const float*)d_in[8];
  const float* W1      = (const float*)d_in[9];
  const float* b1      = (const float*)d_in[10];
  const float* W2      = (const float*)d_in[11];
  const float* b2      = (const float*)d_in[12];
  const float* ln2g    = (const float*)d_in[13];
  const float* ln2b    = (const float*)d_in[14];
  const float* nfg     = (const float*)d_in[15];
  const float* nfb     = (const float*)d_in[16];
  const float* patch_w = (const float*)d_in[17];
  const float* patch_b = (const float*)d_in[18];
  const float* gate_w  = (const float*)d_in[19];
  const float* gate_b  = (const float*)d_in[20];
  float* outp = (float*)d_out;

  char* ws = (char*)d_ws;
  size_t off = 0;
  auto alloc = [&](size_t bytes) {
    void* p = ws + off;
    off += (bytes + 255) & ~(size_t)255;
    return p;
  };
  const size_t R = (size_t)64 * 320;   // 20480 token rows
  float*          xf   = (float*)alloc(R * 256 * 4);
  float*          xnf  = (float*)alloc(R * 256 * 4);
  unsigned short* lnb  = (unsigned short*)alloc(R * 256 * 2);
  unsigned short* qkvb = (unsigned short*)alloc(R * 768 * 2);
  unsigned short* ob   = (unsigned short*)alloc(R * 256 * 2);
  unsigned short* hb   = (unsigned short*)alloc(R * 1024 * 2);
  unsigned short* pA   = (unsigned short*)alloc((size_t)4096 * 768 * 2);
  float*          ptmp = (float*)alloc((size_t)4096 * 256 * 4);
  float*          hcv  = (float*)alloc((size_t)16384 * 256 * 4);
  unsigned short* catb = (unsigned short*)alloc((size_t)16384 * 512 * 2);
  float*          gbuf = (float*)alloc((size_t)16384 * 256 * 4);
  unsigned short* wq_b = (unsigned short*)alloc((size_t)6 * 768 * 256 * 2);
  unsigned short* wo_b = (unsigned short*)alloc((size_t)6 * 256 * 256 * 2);
  unsigned short* w1_b = (unsigned short*)alloc((size_t)6 * 1024 * 256 * 2);
  unsigned short* w2_b = (unsigned short*)alloc((size_t)6 * 256 * 1024 * 2);
  unsigned short* pw_b = (unsigned short*)alloc((size_t)256 * 768 * 2);
  unsigned short* gw_b = (unsigned short*)alloc((size_t)256 * 512 * 2);
  if (off > ws_size) return;

  auto nblk = [](size_t n) { return (unsigned)((n + 255) / 256); };
  auto gemmL = [&](int mode, const unsigned short* A, int lda,
                   const unsigned short* W, int ldw, const float* bias,
                   void* Out, int ldo, int M, int N, int K) {
    int Mb = M / 64, Nb = N / 32;
    unsigned blocks = (unsigned)((Mb * Nb + 7) / 8);
    switch (mode) {
      case 0: gemm_bf16_kernel<0><<<blocks, 256, 0, stream>>>(A, lda, W, ldw, bias, Out, ldo, Mb, Nb, K); break;
      case 1: gemm_bf16_kernel<1><<<blocks, 256, 0, stream>>>(A, lda, W, ldw, bias, Out, ldo, Mb, Nb, K); break;
      case 2: gemm_bf16_kernel<2><<<blocks, 256, 0, stream>>>(A, lda, W, ldw, bias, Out, ldo, Mb, Nb, K); break;
      default:gemm_bf16_kernel<3><<<blocks, 256, 0, stream>>>(A, lda, W, ldw, bias, Out, ldo, Mb, Nb, K); break;
    }
  };

  // Weights -> bf16
  cvt_kernel<<<nblk((size_t)6*768*256), 256, 0, stream>>>(Wqkv, wq_b, (size_t)6*768*256);
  cvt_kernel<<<nblk((size_t)6*256*256), 256, 0, stream>>>(Wo,   wo_b, (size_t)6*256*256);
  cvt_kernel<<<nblk((size_t)6*1024*256),256, 0, stream>>>(W1,   w1_b, (size_t)6*1024*256);
  cvt_kernel<<<nblk((size_t)6*256*1024),256, 0, stream>>>(W2,   w2_b, (size_t)6*256*1024);
  cvt_kernel<<<nblk((size_t)256*768),   256, 0, stream>>>(patch_w, pw_b, (size_t)256*768);
  cvt_kernel<<<nblk((size_t)256*512),   256, 0, stream>>>(gate_w,  gw_b, (size_t)256*512);

  // Token stream + patch embed
  build_tokens_kernel<<<nblk(R * 256), 256, 0, stream>>>(hmap, idx, patch_b, xf, hcv);
  build_patchA_kernel<<<nblk((size_t)4096 * 768), 256, 0, stream>>>(frame, pA);
  gemmL(0, pA, 768, pw_b, 768, nullptr, ptmp, 256, 4096, 256, 768);
  add_patch_kernel<<<nblk((size_t)4096 * 256), 256, 0, stream>>>(ptmp, xf);

  // Transformer layers
  for (int l = 0; l < 6; ++l) {
    ln_kernel<1><<<(unsigned)(R / 8), 256, 0, stream>>>(xf, ln1g + l*256, ln1b + l*256, lnb, (int)R);
    gemmL(2, lnb, 256, wq_b + (size_t)l*768*256, 256, bqkv + l*768, qkvb, 768, (int)R, 768, 256);
    attn_kernel<<<64 * 8 * 10, 256, 0, stream>>>(qkvb, ob);
    gemmL(1, ob, 256, wo_b + (size_t)l*256*256, 256, bo + l*256, xf, 256, (int)R, 256, 256);
    ln_kernel<1><<<(unsigned)(R / 8), 256, 0, stream>>>(xf, ln2g + l*256, ln2b + l*256, lnb, (int)R);
    gemmL(3, lnb, 256, w1_b + (size_t)l*1024*256, 256, b1 + l*1024, hb, 1024, (int)R, 1024, 256);
    gemmL(1, hb, 1024, w2_b + (size_t)l*256*1024, 1024, b2 + l*256, xf, 256, (int)R, 256, 1024);
  }

  // Final LN (f32), gate, blend, scatter
  ln_kernel<0><<<(unsigned)(R / 8), 256, 0, stream>>>(xf, nfg, nfb, xnf, (int)R);
  build_cat_kernel<<<nblk((size_t)16384 * 512), 256, 0, stream>>>(hcv, xnf, catb);
  gemmL(0, catb, 512, gw_b, 512, gate_b, gbuf, 256, 16384, 256, 512);
  copy_kernel<<<nblk((size_t)64 * 1089 * 256), 256, 0, stream>>>(hmap, outp, (size_t)64 * 1089 * 256);
  blend_kernel<<<nblk((size_t)16384 * 256), 256, 0, stream>>>(gbuf, hcv, xnf, idx, outp);
}
